// Block_24876450578627
// MI455X (gfx1250) — matrix operand order repaired and checked by reading, not compiled, not run on hardware
//
#include <hip/hip_runtime.h>
#include <hip/hip_bf16.h>

typedef __attribute__((ext_vector_type(16))) _Float16 v16h;
typedef __attribute__((ext_vector_type(8)))  _Float16 v8h;
typedef __attribute__((ext_vector_type(8)))  float    v8f;

#define N_EMBD 1024
#define N_HEAD 16
#define HEAD_DIM 64
#define BB 8
#define TT 1024
#define II 257

// ---------------------------------------------------------------------------
// CDNA5 async memory->LDS copy (16B per lane), tracked by ASYNCcnt.
// VDST = VGPR holding the LDS byte address; VADDR = 64-bit global address.
// ---------------------------------------------------------------------------
__device__ __forceinline__ void async_ld16(void* lds, const void* gaddr) {
  asm volatile("global_load_async_to_lds_b128 %0, %1, off"
               :
               : "v"((unsigned)(unsigned long long)lds),
                 "v"((unsigned long long)gaddr)
               : "memory");
}
__device__ __forceinline__ void async_wait0() {
  asm volatile("s_wait_asynccnt 0x0" ::: "memory");
}

// ---------------------------------------------------------------------------
// Weight cast + transpose: W[K,N] f32 -> Wt[N,K] f16 (32x32 tiles via LDS).
// ---------------------------------------------------------------------------
__global__ void castT_kernel(const float* __restrict__ W,
                             _Float16* __restrict__ Wt, int K, int N) {
  __shared__ float tile[32][33];
  const int k0 = blockIdx.x * 32, n0 = blockIdx.y * 32;
  const int tx = threadIdx.x & 31, ty = threadIdx.x >> 5;  // 256 thr = 32x8
#pragma unroll
  for (int r = ty; r < 32; r += 8)
    tile[r][tx] = W[(size_t)(k0 + r) * N + n0 + tx];
  __syncthreads();
#pragma unroll
  for (int r = ty; r < 32; r += 8)
    Wt[(size_t)(n0 + r) * K + k0 + tx] = (_Float16)tile[tx][r];
}

// Elementwise f32 -> f16 cast (encoder activations).
__global__ void cast_kernel(const float* __restrict__ in,
                            _Float16* __restrict__ out, int n) {
  const int i = blockIdx.x * 256 + threadIdx.x;
  if (i < n) out[i] = (_Float16)in[i];
}

// ---------------------------------------------------------------------------
// LayerNorm: f32 in, f16 out. One 256-thread block per row of 1024.
// ---------------------------------------------------------------------------
__global__ void ln_kernel(const float* __restrict__ x,
                          const float* __restrict__ g,
                          const float* __restrict__ b,
                          _Float16* __restrict__ out) {
  const int row = blockIdx.x;
  const float* xr = x + (size_t)row * N_EMBD;
  _Float16* outr = out + (size_t)row * N_EMBD;
  const int tid = threadIdx.x;

  float v[4];
  float s = 0.f, s2 = 0.f;
#pragma unroll
  for (int j = 0; j < 4; ++j) {
    v[j] = xr[tid + j * 256];
    s += v[j];
    s2 += v[j] * v[j];
  }
#pragma unroll
  for (int off = 16; off > 0; off >>= 1) {
    s  += __shfl_xor(s,  off, 32);
    s2 += __shfl_xor(s2, off, 32);
  }
  __shared__ float red[16];
  const int wid = tid >> 5;
  if ((tid & 31) == 0) { red[wid] = s; red[8 + wid] = s2; }
  __syncthreads();
  float ts = 0.f, ts2 = 0.f;
#pragma unroll
  for (int w = 0; w < 8; ++w) { ts += red[w]; ts2 += red[8 + w]; }
  const float mean = ts * (1.0f / N_EMBD);
  const float var  = ts2 * (1.0f / N_EMBD) - mean * mean;
  const float rstd = rsqrtf(var + 1e-5f);
#pragma unroll
  for (int j = 0; j < 4; ++j) {
    const int c = tid + j * 256;
    outr[c] = (_Float16)((v[j] - mean) * rstd * g[c] + b[c]);
  }
}

// ---------------------------------------------------------------------------
// WMMA GEMM: out[M,N] = A[M,K](f16) @ Wt[N,K](f16)^T + bias (+GELU)(+res).
// 128x64 tile, 128 threads = 4 waves; each wave 32 rows x 64 cols = 8 accums.
// Double-buffered LDS; in-bounds tiles staged with async copies (ASYNCcnt)
// overlapping the WMMA stream.
// ---------------------------------------------------------------------------
template <bool GELU, bool RES, bool BOUND_M, bool OUTH>
__global__ void gemm_wmma(const _Float16* __restrict__ A,
                          const _Float16* __restrict__ Wg,
                          const float* __restrict__ bias,
                          const float* __restrict__ res,
                          void* __restrict__ outp,
                          int M, int N, int K) {
  const int m0 = blockIdx.y * 128;
  const int n0 = blockIdx.x * 64;

  __shared__ _Float16 As[2][128][40];  // [m][k]
  __shared__ _Float16 Wt[2][64][40];   // [n][k]

  const int tid  = threadIdx.x;        // 128
  const int lane = tid & 31;
  const int wave = tid >> 5;
  const int hi   = lane >> 4;
  const int ln   = lane & 15;

  v8f acc[2][4] = {};

  auto stage = [&](int buf, int k0) {
    // A tile 128x32 halfs: 512 v8h segs, 4 per thread.
    if (BOUND_M) {
      v8h vals[4];
#pragma unroll
      for (int j = 0; j < 4; ++j) {
        const int seg = tid * 4 + j;
        const int r = seg >> 2, coff = (seg & 3) * 8;
        vals[j] = (v8h){};
        if ((m0 + r) < M)
          vals[j] = *(const v8h*)(A + (size_t)(m0 + r) * K + k0 + coff);
      }
#pragma unroll
      for (int j = 0; j < 4; ++j) {
        const int seg = tid * 4 + j;
        const int r = seg >> 2, coff = (seg & 3) * 8;
        *(v8h*)&As[buf][r][coff] = vals[j];
      }
    } else {
#pragma unroll
      for (int j = 0; j < 4; ++j) {
        const int seg = tid * 4 + j;
        const int r = seg >> 2, coff = (seg & 3) * 8;
        async_ld16(&As[buf][r][coff],
                   A + (size_t)(m0 + r) * K + k0 + coff);
      }
    }
    // Wt tile 64x32 halfs: 256 v8h segs, 2 per thread (always in bounds).
#pragma unroll
    for (int j = 0; j < 2; ++j) {
      const int seg = tid * 2 + j;
      const int r = seg >> 2, coff = (seg & 3) * 8;
      async_ld16(&Wt[buf][r][coff],
                 Wg + (size_t)(n0 + r) * K + k0 + coff);
    }
  };

  stage(0, 0);
  async_wait0();
  __syncthreads();

  int cur = 0;
  for (int k0 = 0; k0 < K; k0 += 32) {
    const int nxt = cur ^ 1;
    if (k0 + 32 < K) stage(nxt, k0 + 32);  // async; overlaps compute below

    v16h af[2];
#pragma unroll
    for (int u = 0; u < 2; ++u) {
      const int ar = wave * 32 + u * 16 + ln;
#pragma unroll
      for (int j = 0; j < 8; ++j) {
        af[u][j]     = As[cur][ar][hi * 8 + j];
        af[u][8 + j] = As[cur][ar][16 + hi * 8 + j];
      }
    }
#pragma unroll
    for (int t = 0; t < 4; ++t) {
      v16h bf;
      const int bn = t * 16 + ln;
#pragma unroll
      for (int j = 0; j < 16; ++j) bf[j] = Wt[cur][bn][hi * 8 + (j & 7) + ((j >> 3) << 4)];
#pragma unroll
      for (int u = 0; u < 2; ++u)
        acc[u][t] = __builtin_amdgcn_wmma_f32_16x16x32_f16(
            false, af[u], false, bf, (short)0, acc[u][t], false, false);
    }

    async_wait0();     // next tile landed in LDS (wave-local ASYNCcnt)
    __syncthreads();   // all waves done with cur + staged nxt
    cur = nxt;
  }

  // Epilogue.
#pragma unroll
  for (int t = 0; t < 4; ++t) {
    const int col = n0 + t * 16 + ln;
    const float bv = bias[col];
#pragma unroll
    for (int u = 0; u < 2; ++u) {
#pragma unroll
      for (int i = 0; i < 8; ++i) {
        const int row = m0 + wave * 32 + u * 16 + i + 8 * hi;
        if (BOUND_M && row >= M) continue;
        float v = acc[u][t][i] + bv;
        if (GELU) {
          const float uu = v;
          const float c = 0.7978845608028654f * (uu + 0.044715f * uu * uu * uu);
          v = 0.5f * uu * (1.0f + tanhf(c));
        }
        if (RES) v += res[(size_t)row * N + col];
        if (OUTH)
          ((_Float16*)outp)[(size_t)row * N + col] = (_Float16)v;
        else
          ((float*)outp)[(size_t)row * N + col] = v;
      }
    }
  }
}

// ---------------------------------------------------------------------------
// Flash attention (WMMA), f16 in / f16 out. Block = (b, h, 64 Q rows),
// 4 waves, each wave owns a 16-row strip; online softmax is wave-local.
// CAUSAL instantiation (kvLen = T, always full tiles) stages K with async
// copies; cross variant (kvLen = 257) uses bounded register staging.
// ---------------------------------------------------------------------------
template <bool CAUSAL>
__global__ void attn_kernel(const _Float16* __restrict__ Qp,
                            const _Float16* __restrict__ Kp,
                            const _Float16* __restrict__ Vp,
                            _Float16* __restrict__ Op,
                            int qStride, int kStride, int vStride, int oStride,
                            long long qBatch, long long kBatch,
                            long long vBatch, long long oBatch,
                            int kvLen, float scale) {
  const int h  = blockIdx.y;
  const int bz = blockIdx.z;
  const int q0 = blockIdx.x * 64;

  const _Float16* Qg = Qp + (long long)bz * qBatch + h * HEAD_DIM;
  const _Float16* Kg = Kp + (long long)bz * kBatch + h * HEAD_DIM;
  const _Float16* Vg = Vp + (long long)bz * vBatch + h * HEAD_DIM;
  _Float16*       Og = Op + (long long)bz * oBatch + h * HEAD_DIM;

  __shared__ _Float16 Qs[64][72];
  __shared__ _Float16 Ks[32][72];
  __shared__ _Float16 Vt[64][40];     // [d][s]
  __shared__ _Float16 Ps[4][16][32];

  const int tid  = threadIdx.x;       // 128
  const int lane = tid & 31;
  const int wave = tid >> 5;
  const int hi   = lane >> 4;
  const int ln   = lane & 15;

  // Stage Q 64x64 halfs via async copies (always in bounds).
#pragma unroll
  for (int j = 0; j < 4; ++j) {
    const int seg = tid * 4 + j;
    const int r = seg >> 3, coff = (seg & 7) * 8;
    async_ld16(&Qs[r][coff], Qg + (size_t)(q0 + r) * qStride + coff);
  }
  async_wait0();
  __syncthreads();

  v16h aq[2];
#pragma unroll
  for (int kc = 0; kc < 2; ++kc) {
#pragma unroll
    for (int j = 0; j < 8; ++j) {
      aq[kc][j]     = Qs[wave * 16 + ln][kc * 32 + hi * 8 + j];
      aq[kc][8 + j] = Qs[wave * 16 + ln][kc * 32 + 16 + hi * 8 + j];
    }
  }

  v8f o[4] = {};
  float m[8], l[8];
#pragma unroll
  for (int i = 0; i < 8; ++i) { m[i] = -1e30f; l[i] = 0.f; }

  const int sEnd = CAUSAL ? min(kvLen, q0 + 64) : kvLen;
  for (int s0 = 0; s0 < sEnd; s0 += 32) {
    __syncthreads();
    // Stage K rows.
    if (CAUSAL) {
#pragma unroll
      for (int j = 0; j < 2; ++j) {
        const int seg = tid * 2 + j;
        const int r = seg >> 3, coff = (seg & 7) * 8;
        async_ld16(&Ks[r][coff], Kg + (size_t)(s0 + r) * kStride + coff);
      }
    } else {
      v8h kv[2];
#pragma unroll
      for (int j = 0; j < 2; ++j) {
        const int seg = tid * 2 + j;
        const int r = seg >> 3, coff = (seg & 7) * 8;
        kv[j] = (v8h){};
        if (s0 + r < kvLen)
          kv[j] = *(const v8h*)(Kg + (size_t)(s0 + r) * kStride + coff);
      }
#pragma unroll
      for (int j = 0; j < 2; ++j) {
        const int seg = tid * 2 + j;
        const int r = seg >> 3, coff = (seg & 7) * 8;
        *(v8h*)&Ks[r][coff] = kv[j];
      }
    }
    // Stage V transposed (register path; needs layout change).
    {
      const int r = tid >> 2, c = (tid & 3) * 16;
      v8h v0 = {}, v1 = {};
      if (CAUSAL || (s0 + r < kvLen)) {
        const _Float16* vs = Vg + (size_t)(s0 + r) * vStride + c;
        v0 = *(const v8h*)vs;
        v1 = *(const v8h*)(vs + 8);
      }
#pragma unroll
      for (int j = 0; j < 8; ++j) {
        Vt[c + j][r]     = v0[j];
        Vt[c + 8 + j][r] = v1[j];
      }
    }
    if (s0 + 32 < sEnd) {
      __builtin_prefetch(Kg + (size_t)(s0 + 32 + (tid >> 2)) * kStride, 0, 3);
      __builtin_prefetch(Vg + (size_t)(s0 + 32 + (tid >> 2)) * vStride, 0, 3);
    }
    if (CAUSAL) async_wait0();
    __syncthreads();

    // S = Q K^T (16x32 strip per wave).
    v8f sacc[2] = {};
#pragma unroll
    for (int st = 0; st < 2; ++st) {
#pragma unroll
      for (int kc = 0; kc < 2; ++kc) {
        v16h bf;
        const int bn = st * 16 + ln;
#pragma unroll
        for (int j = 0; j < 16; ++j) bf[j] = Ks[bn][kc * 32 + hi * 8 + (j & 7) + ((j >> 3) << 4)];
        sacc[st] = __builtin_amdgcn_wmma_f32_16x16x32_f16(
            false, aq[kc], false, bf, (short)0, sacc[st], false, false);
      }
    }

    // Online softmax.
#pragma unroll
    for (int i = 0; i < 8; ++i) {
      const int rloc = wave * 16 + i + 8 * hi;
      float s0v = sacc[0][i] * scale;
      float s1v = sacc[1][i] * scale;
      const int c0 = s0 + ln, c1 = s0 + 16 + ln;
      bool ok0 = c0 < kvLen, ok1 = c1 < kvLen;
      if (CAUSAL) {
        const int rg = q0 + rloc;
        ok0 = ok0 && (c0 <= rg);
        ok1 = ok1 && (c1 <= rg);
      }
      if (!ok0) s0v = -1e30f;
      if (!ok1) s1v = -1e30f;
      float mcur = fmaxf(s0v, s1v);
#pragma unroll
      for (int off = 1; off < 16; off <<= 1)
        mcur = fmaxf(mcur, __shfl_xor(mcur, off, 32));
      const float mnew = fmaxf(m[i], mcur);
      const float p0 = __expf(s0v - mnew);
      const float p1 = __expf(s1v - mnew);
      float ps = p0 + p1;
#pragma unroll
      for (int off = 1; off < 16; off <<= 1) ps += __shfl_xor(ps, off, 32);
      const float alpha = __expf(m[i] - mnew);
      l[i] = l[i] * alpha + ps;
      m[i] = mnew;
#pragma unroll
      for (int t = 0; t < 4; ++t) o[t][i] = o[t][i] * alpha;
      Ps[wave][i + 8 * hi][ln]      = (_Float16)p0;
      Ps[wave][i + 8 * hi][16 + ln] = (_Float16)p1;
    }

    // O += P V.
    v16h pf;
#pragma unroll
    for (int j = 0; j < 8; ++j) {
      pf[j]     = Ps[wave][ln][hi * 8 + j];
      pf[8 + j] = Ps[wave][ln][16 + hi * 8 + j];
    }
#pragma unroll
    for (int t = 0; t < 4; ++t) {
      v16h vf;
      const int vn = t * 16 + ln;
#pragma unroll
      for (int j = 0; j < 16; ++j) vf[j] = Vt[vn][hi * 8 + (j & 7) + ((j >> 3) << 4)];
      o[t] = __builtin_amdgcn_wmma_f32_16x16x32_f16(
          false, pf, false, vf, (short)0, o[t], false, false);
    }
  }

#pragma unroll
  for (int t = 0; t < 4; ++t) {
#pragma unroll
    for (int i = 0; i < 8; ++i) {
      const int rg = q0 + wave * 16 + i + 8 * hi;
      const float inv = 1.0f / l[i];
      Og[(size_t)rg * oStride + t * 16 + ln] = (_Float16)(o[t][i] * inv);
    }
  }
}

// ---------------------------------------------------------------------------
// Orchestration
// ---------------------------------------------------------------------------
extern "C" void kernel_launch(void* const* d_in, const int* in_sizes, int n_in,
                              void* d_out, int out_size, void* d_ws, size_t ws_size,
                              hipStream_t stream) {
  const float* x    = (const float*)d_in[0];
  const float* enc  = (const float*)d_in[1];
  const float* ln1g = (const float*)d_in[2];
  const float* ln1b = (const float*)d_in[3];
  const float* ln2g = (const float*)d_in[4];
  const float* ln2b = (const float*)d_in[5];
  const float* ln3g = (const float*)d_in[6];
  const float* ln3b = (const float*)d_in[7];
  const float* w_qkv = (const float*)d_in[8];
  const float* b_qkv = (const float*)d_in[9];
  const float* w_ao  = (const float*)d_in[10];
  const float* b_ao  = (const float*)d_in[11];
  const float* w_q   = (const float*)d_in[12];
  const float* b_q   = (const float*)d_in[13];
  const float* w_kv  = (const float*)d_in[14];
  const float* b_kv  = (const float*)d_in[15];
  const float* w_co  = (const float*)d_in[16];
  const float* b_co  = (const float*)d_in[17];
  const float* w_fc  = (const float*)d_in[18];
  const float* b_fc  = (const float*)d_in[19];
  const float* w_mo  = (const float*)d_in[20];
  const float* b_mo  = (const float*)d_in[21];
  float* out = (float*)d_out;

  const size_t BT = (size_t)BB * TT;            // 8192
  const size_t C  = N_EMBD;
  const int    NI = BB * II * N_EMBD;

  _Float16* p = (_Float16*)d_ws;
  _Float16* wt_qkv = p; p += C * 3 * C;
  _Float16* wt_ao  = p; p += C * C;
  _Float16* wt_q   = p; p += C * C;
  _Float16* wt_kv  = p; p += C * 2 * C;
  _Float16* wt_co  = p; p += C * C;
  _Float16* wt_fc  = p; p += C * 4 * C;
  _Float16* wt_mo  = p; p += 4 * C * C;
  _Float16* enc_h  = p; p += (size_t)NI;
  _Float16* R0h    = p; p += BT * C;            // LN out / cross-attn out
  _Float16* R1h    = p; p += BT * 4 * C;        // qkv / kv / fc
  _Float16* R2h    = p; p += BT * C;            // self-attn out / q
  float* R3 = (float*)p;                        // x1
  float* R4 = R3 + BT * C;                      // x2

  const float scale = 0.125f;                   // 1/sqrt(64)
  const dim3 blk128(128), blk256(256);
  const dim3 gAttn(TT / 64, N_HEAD, BB);

  // ---- weight / encoder precasts (f32 -> f16, weights transposed) ----
  castT_kernel<<<dim3(C / 32, 3 * C / 32), blk256, 0, stream>>>(w_qkv, wt_qkv, C, 3 * C);
  castT_kernel<<<dim3(C / 32, C / 32),     blk256, 0, stream>>>(w_ao,  wt_ao,  C, C);
  castT_kernel<<<dim3(C / 32, C / 32),     blk256, 0, stream>>>(w_q,   wt_q,   C, C);
  castT_kernel<<<dim3(C / 32, 2 * C / 32), blk256, 0, stream>>>(w_kv,  wt_kv,  C, 2 * C);
  castT_kernel<<<dim3(C / 32, C / 32),     blk256, 0, stream>>>(w_co,  wt_co,  C, C);
  castT_kernel<<<dim3(C / 32, 4 * C / 32), blk256, 0, stream>>>(w_fc,  wt_fc,  C, 4 * C);
  castT_kernel<<<dim3(4 * C / 32, C / 32), blk256, 0, stream>>>(w_mo,  wt_mo,  4 * C, C);
  cast_kernel<<<dim3((NI + 255) / 256), blk256, 0, stream>>>(enc, enc_h, NI);

  // ---- self-attention ----
  ln_kernel<<<dim3(BT), blk256, 0, stream>>>(x, ln1g, ln1b, R0h);
  gemm_wmma<false, false, false, true><<<dim3(3 * C / 64, BT / 128), blk128, 0, stream>>>(
      R0h, wt_qkv, b_qkv, nullptr, R1h, (int)BT, 3 * C, C);
  attn_kernel<true><<<gAttn, blk128, 0, stream>>>(
      R1h, R1h + C, R1h + 2 * C, R2h,
      3 * C, 3 * C, 3 * C, C,
      (long long)TT * 3 * C, (long long)TT * 3 * C,
      (long long)TT * 3 * C, (long long)TT * C,
      TT, scale);
  gemm_wmma<false, true, false, false><<<dim3(C / 64, BT / 128), blk128, 0, stream>>>(
      R2h, wt_ao, b_ao, x, R3, (int)BT, C, C);

  // ---- cross-attention ----
  ln_kernel<<<dim3(BT), blk256, 0, stream>>>(R3, ln2g, ln2b, R0h);
  gemm_wmma<false, false, false, true><<<dim3(C / 64, BT / 128), blk128, 0, stream>>>(
      R0h, wt_q, b_q, nullptr, R2h, (int)BT, C, C);
  const int MI = BB * II;                        // 2056
  gemm_wmma<false, false, true, true><<<dim3(2 * C / 64, (MI + 127) / 128), blk128, 0, stream>>>(
      enc_h, wt_kv, b_kv, nullptr, R1h, MI, 2 * C, C);
  attn_kernel<false><<<gAttn, blk128, 0, stream>>>(
      R2h, R1h, R1h + C, R0h,
      C, 2 * C, 2 * C, C,
      (long long)TT * C, (long long)II * 2 * C,
      (long long)II * 2 * C, (long long)TT * C,
      II, scale);
  gemm_wmma<false, true, false, false><<<dim3(C / 64, BT / 128), blk128, 0, stream>>>(
      R0h, wt_co, b_co, R3, R4, (int)BT, C, C);

  // ---- MLP ----
  ln_kernel<<<dim3(BT), blk256, 0, stream>>>(R4, ln3g, ln3b, R0h);
  gemm_wmma<true, false, false, true><<<dim3(4 * C / 64, BT / 128), blk128, 0, stream>>>(
      R0h, wt_fc, b_fc, nullptr, R1h, (int)BT, 4 * C, C);
  gemm_wmma<false, true, false, false><<<dim3(C / 64, BT / 128), blk128, 0, stream>>>(
      R1h, wt_mo, b_mo, R4, out, (int)BT, C, 4 * C);
}
